// VADPredictor_8169027797084
// MI455X (gfx1250) — compile-verified, matrix operand order checked by reading, not run on hardware
//
#include <hip/hip_runtime.h>

// ---------------------------------------------------------------------------
// VAD 2-layer autoregressive GRU for MI455X (gfx1250), wave32 + WMMA bf16.
// Persistent 8-workgroup kernel; recurrent weights LDS-resident (loaded once
// via TDM tensor_load_to_lds); h state exchanged through L2 with a monotonic
// atomic grid barrier (2 per step) and refreshed into LDS with
// global_load_async_to_lds_b128 (ASYNCcnt).
// ---------------------------------------------------------------------------

#define IDIM   512
#define NCH    256
#define ODIM   3
#define BATCH  64
#define SEQ    1024
#define NWG    8      // persistent workgroups
#define CPW    32     // channels per workgroup
#define NTW    6      // N-tiles of 16 per WG (96 gate cols: [r|z|n] x 32ch)

typedef unsigned short u16;
typedef unsigned int   u32;
typedef unsigned long long u64;
typedef __attribute__((ext_vector_type(16))) __bf16 v16bf;
typedef __attribute__((ext_vector_type(8)))  float  v8f;
typedef __attribute__((ext_vector_type(4)))  int    v4i;
typedef __attribute__((ext_vector_type(8)))  int    v8i;

// ---- workspace layout (bytes) ----------------------------------------------
#define WS_W0X   ((size_t)0)
#define SZ_W0X   ((size_t)NWG*NTW*16*512*2)          // 786,432
#define WS_WH    (WS_W0X + SZ_W0X)
#define SZ_WH    ((size_t)3*NWG*NTW*8*512*2)         // 1,179,648
#define WS_BA0   (WS_WH + SZ_WH)                     // 8*96 f32
#define WS_BH0   (WS_BA0 + NWG*96*4)                 // 8*32 f32
#define WS_BA1   (WS_BH0 + NWG*32*4)
#define WS_BH1   (WS_BA1 + NWG*96*4)
#define WS_WC    (WS_BH1 + NWG*32*4)                 // 8*96*3 f32
#define WS_PW    (WS_WC  + (size_t)NWG*96*3*4)       // 3*256 f32
#define WS_PB    (WS_PW  + 3*256*4)                  // 3 f32 (pad 64)
#define WS_H0    (WS_PB  + 64)                       // 64*256 bf16
#define WS_H1    (WS_H0  + (size_t)BATCH*NCH*2)
#define WS_BAR   (WS_H1  + (size_t)BATCH*NCH*2)      // barrier counter
#define WS_TOTAL (WS_BAR + 64)

// ---- LDS layout (bytes), carved from dynamic shared ------------------------
#define L_XT    0                         // x_t  bf16 [64][520] (pad vs banks)
#define L_H0    66560                     // h0   bf16 [64][264]
#define L_H1    100352                    // h1   bf16 [64][264]
#define L_WH    134144                    // 3 * 24576 halves of Wh fragments
#define L_VP    281600                    // prev_v f32 [64][4]
#define L_SB    282624                    // biases+Wc: 544 f32
#define L_PW    284800                    // proj_w 768 f32 + proj_b
#define SMEM_BYTES 287904

__device__ __forceinline__ u16 f2bf(float f) {
    u32 u = __float_as_uint(f);
    u32 r = u + 0x7fffu + ((u >> 16) & 1u);
    return (u16)(r >> 16);
}
__device__ __forceinline__ float bf2f(u16 h) {
    return __uint_as_float(((u32)h) << 16);
}
__device__ __forceinline__ float sigm(float x) {
    return 1.0f / (1.0f + __expf(-x));
}
__device__ __forceinline__ float fast_tanh(float x) {
#if __has_builtin(__builtin_amdgcn_tanhf)
    return __builtin_amdgcn_tanhf(x);     // v_tanh_f32 (gfx1250 trans op)
#else
    return tanhf(x);
#endif
}
__device__ __forceinline__ v8f splat8(float v) {
    v8f r;
#pragma unroll
    for (int i = 0; i < 8; ++i) r[i] = v;
    return r;
}
__device__ __forceinline__ v8f wmma_bf16(v16bf a, v16bf b, v8f c) {
    return __builtin_amdgcn_wmma_f32_16x16x32_bf16(false, a, false, b,
                                                   (short)0, c, false, false);
}

// A fragment (16x32 bf16, documented CDNA5 layout) gathered from LDS.
__device__ __forceinline__ v16bf loadA(const u16* base, int stride, int m0, int k0) {
    int l = threadIdx.x & 31;
    const u16* p = base + (m0 + (l & 15)) * stride + k0 + ((l & 16) ? 8 : 0);
    union { u32 u[8]; v16bf v; } U;
    U.u[0] = *(const u32*)(p + 0);
    U.u[1] = *(const u32*)(p + 2);
    U.u[2] = *(const u32*)(p + 4);
    U.u[3] = *(const u32*)(p + 6);
    U.u[4] = *(const u32*)(p + 16);
    U.u[5] = *(const u32*)(p + 18);
    U.u[6] = *(const u32*)(p + 20);
    U.u[7] = *(const u32*)(p + 22);
    return U.v;
}
// B fragment: pre-swizzled contiguous 1KB blob, 32B per lane (2x b128 loads).
__device__ __forceinline__ v16bf loadB(const u16* fragbase) {
    int l = threadIdx.x & 31;
    const uint4* p = (const uint4*)(fragbase + l * 16);
    union { uint4 q[2]; v16bf v; } U;
    U.q[0] = p[0];
    U.q[1] = p[1];
    return U.v;
}

// async memory -> LDS, 16 bytes per active lane (ASYNCcnt tracked)
__device__ __forceinline__ void async_g2l_b128(u32 lds_addr, const void* gptr) {
    asm volatile("global_load_async_to_lds_b128 %0, %1, off"
                 :: "v"(lds_addr), "v"((u64)(size_t)gptr)
                 : "memory");
}
__device__ __forceinline__ void wait_async0() {
#if __has_builtin(__builtin_amdgcn_s_wait_asynccnt)
    __builtin_amdgcn_s_wait_asynccnt(0);
#else
    asm volatile("s_wait_asynccnt 0x0" ::: "memory");
#endif
}

// monotonic grid barrier over NWG blocks
__device__ __forceinline__ void gbar(int* bar, int target) {
    __threadfence();
    __syncthreads();
    if (threadIdx.x == 0) {
        __hip_atomic_fetch_add(bar, 1, __ATOMIC_ACQ_REL, __HIP_MEMORY_SCOPE_AGENT);
        while (__hip_atomic_load(bar, __ATOMIC_ACQUIRE, __HIP_MEMORY_SCOPE_AGENT) < target)
            __builtin_amdgcn_s_sleep(1);
    }
    __syncthreads();
    __threadfence();
}

// ---------------------------------------------------------------------------
__global__ void vad_init(int* bar) {
    if (threadIdx.x < 16) bar[threadIdx.x] = 0;
}

// bias folding, composite Wc = w_ih0[:,512:] @ embed_w, proj copy
__global__ void vad_pack_small(const float* __restrict__ w_ih0,
                               const float* __restrict__ embed_w,
                               const float* __restrict__ embed_b,
                               const float* __restrict__ b_ih0,
                               const float* __restrict__ b_hh0,
                               const float* __restrict__ b_ih1,
                               const float* __restrict__ b_hh1,
                               const float* __restrict__ proj_w,
                               const float* __restrict__ proj_b,
                               float* ba0, float* bh0, float* ba1, float* bh1,
                               float* wc, float* pw, float* pb) {
    int gt = blockIdx.x * blockDim.x + threadIdx.x;
    if (gt >= 768) return;
    int row = gt, gate = row >> 8, ch = row & 255;
    int wg = ch >> 5, j = (gate << 5) + (ch & 31);
    float c0 = 0.f, c1 = 0.f, c2 = 0.f, ef = 0.f;
    const float* wrow = w_ih0 + (size_t)row * (IDIM + NCH) + IDIM;
    for (int e = 0; e < NCH; ++e) {
        float w = wrow[e];
        c0 += w * embed_w[e * 3 + 0];
        c1 += w * embed_w[e * 3 + 1];
        c2 += w * embed_w[e * 3 + 2];
        ef += w * embed_b[e];
    }
    float* wcp = wc + ((size_t)wg * 96 + j) * 3;
    wcp[0] = c0; wcp[1] = c1; wcp[2] = c2;
    if (gate < 2) {
        ba0[wg * 96 + j] = b_ih0[row] + b_hh0[row] + ef;
        ba1[wg * 96 + j] = b_ih1[row] + b_hh1[row];
    } else {
        ba0[wg * 96 + j] = b_ih0[row] + ef;       // i_n part
        bh0[wg * 32 + (ch & 31)] = b_hh0[row];    // h_n part
        ba1[wg * 96 + j] = b_ih1[row];
        bh1[wg * 32 + (ch & 31)] = b_hh1[row];
    }
    pw[gt] = proj_w[gt];
    if (gt < 3) pb[gt] = proj_b[gt];
}

// mapping of one bf16 half inside a B fragment -> (k_local, n_local)
__device__ __forceinline__ void bmap(int lane, int vh, int& k_local, int& n_local) {
    n_local = lane & 15;
    int kb = (lane >> 4) * 8;
    k_local = ((vh >> 3) << 4) + kb + (vh & 7);
}

__global__ void vad_pack_wx(const float* __restrict__ w_ih0, u16* dst) {
    int idx = blockIdx.x * blockDim.x + threadIdx.x;   // < 786432
    int f = idx >> 9, within = idx & 511;
    int lane = within >> 4, vh = within & 15;
    int ks = f & 15, nt = (f >> 4) % NTW, wg = f / (NTW * 16);
    int kl, nl; bmap(lane, vh, kl, nl);
    int j = nt * 16 + nl, gate = j >> 5, ch = wg * CPW + (j & 31);
    int row = (gate << 8) + ch;
    int kg = ks * 32 + kl;
    dst[idx] = f2bf(w_ih0[(size_t)row * (IDIM + NCH) + kg]);
}

__global__ void vad_pack_wh(const float* __restrict__ w_hh0,
                            const float* __restrict__ w_ih1,
                            const float* __restrict__ w_hh1, u16* dst) {
    int idx = blockIdx.x * blockDim.x + threadIdx.x;   // < 589824
    int f = idx >> 9, within = idx & 511;
    int lane = within >> 4, vh = within & 15;
    int ks = f & 7, nt = (f >> 3) % NTW, wg = (f / (NTW * 8)) & 7, mi = f / (NWG * NTW * 8);
    int kl, nl; bmap(lane, vh, kl, nl);
    int j = nt * 16 + nl, gate = j >> 5, ch = wg * CPW + (j & 31);
    int row = (gate << 8) + ch;
    int kg = ks * 32 + kl;
    const float* src = (mi == 0) ? w_hh0 : (mi == 1) ? w_ih1 : w_hh1;
    dst[idx] = f2bf(src[(size_t)row * NCH + kg]);
}

// ---------------------------------------------------------------------------
__global__ void __launch_bounds__(256, 1)
vad_rnn(const float* __restrict__ x, const u16* __restrict__ wxp,
        const u16* __restrict__ whp,
        const float* __restrict__ ba0, const float* __restrict__ bh0,
        const float* __restrict__ ba1, const float* __restrict__ bh1,
        const float* __restrict__ wc,  const float* __restrict__ pw,
        const float* __restrict__ pb,
        u16* h0buf, u16* h1buf, int* bar, float* __restrict__ out) {
    extern __shared__ char smem[];
    u16*   xt   = (u16*)(smem + L_XT);    // [64][520]
    u16*   h0f  = (u16*)(smem + L_H0);    // [64][264]
    u16*   h1f  = (u16*)(smem + L_H1);    // [64][264]
    u16*   wlds = (u16*)(smem + L_WH);    // [3][6][8][512]
    float* vp   = (float*)(smem + L_VP);  // [64][4]
    float* sb   = (float*)(smem + L_SB);  // 544 f32
    float* spw  = (float*)(smem + L_PW);  // 768 + 3 f32

    const int wg = blockIdx.x, tid = threadIdx.x;
    const int lane = tid & 31, wv = tid >> 5;
    const int mt = wv >> 1, ct = wv & 1;      // M tile 0..3, channel subtile 0..1
    const int jr = ct * 16 + (lane & 15);     // 0..31 within this WG's slice
    const int cg = wg * CPW + jr;             // global channel

    // --- one-shot TDM: pull this WG's 144KB recurrent-weight block into LDS.
    // 2D descriptor: 3 rows of 24576 bf16, global row stride 196608 elements.
    if (wv == 0) {
        u64 ga = (u64)(size_t)(whp + (size_t)wg * 24576);
        u32 la = (u32)(size_t)wlds;
        v4i g0; v8i g1;
        g0[0] = 1;                                   // count = 1 (valid D#)
        g0[1] = (int)la;                             // lds_addr
        g0[2] = (int)(u32)ga;                        // global_addr[31:0]
        g0[3] = (int)(((u32)(ga >> 32) & 0x01FFFFFFu) | 0x80000000u); // +type=2
        g1[0] = 1 << 16;                             // data_size = 2 bytes
        g1[1] = (int)0x60000000u;                    // tensor_dim0 = 24576
        g1[2] = 0x00030000;                          // tensor_dim1 = 3
        g1[3] = (int)0x60000000u;                    // tile_dim0 = 24576
        g1[4] = 3;                                   // tile_dim1 = 3
        g1[5] = 0x00030000;                          // dim0_stride = 196608
        g1[6] = 0; g1[7] = 0;
        asm volatile("tensor_load_to_lds %0, %1" :: "s"(g0), "s"(g1) : "memory");
        __builtin_amdgcn_s_wait_tensorcnt(0);
    }

    for (int r = tid; r < 96; r += 256) { sb[r] = ba0[wg * 96 + r]; sb[128 + r] = ba1[wg * 96 + r]; }
    if (tid < 32) { sb[96 + tid] = bh0[wg * 32 + tid]; sb[224 + tid] = bh1[wg * 32 + tid]; }
    for (int r = tid; r < 288; r += 256) sb[256 + r] = wc[wg * 288 + r];
    for (int r = tid; r < 768; r += 256) spw[r] = pw[r];
    if (tid < 3) spw[768 + tid] = pb[tid];
    for (int r = tid; r < BATCH * 264; r += 256) { h0f[r] = 0; h1f[r] = 0; }
    if (tid < 256) vp[tid] = 0.f;
    __syncthreads();

    const u32 l_h0 = (u32)(size_t)h0f;
    const u32 l_h1 = (u32)(size_t)h1f;

    for (int t = 0; t < SEQ; ++t) {
        // ---- stage x_t into LDS (fp32 -> bf16), prefetch x_{t+1} ----------
        for (int r = tid; r < BATCH * IDIM; r += 256) {
            int m = r >> 9, i = r & 511;
            const float* gp = &x[((size_t)m * SEQ + t) * IDIM + i];
            xt[m * 520 + i] = f2bf(*gp);
            if ((r & 63) == 0 && t + 1 < SEQ)
                __builtin_prefetch(gp + IDIM, 0, 0);  // global_prefetch_b8
        }
        __syncthreads();

        // ============================ layer 0 ==============================
        v8f ar  = splat8(sb[jr]);         // r (gi+gh combined)
        v8f az  = splat8(sb[32 + jr]);    // z (combined)
        v8f ain = splat8(sb[64 + jr]);    // i_n (+ emb-bias fold)
        v8f ahn = splat8(sb[96 + jr]);    // h_n
        // folded embedding feedback: gi += prev_v @ Wc^T  (3 FMAs / element)
#pragma unroll
        for (int i = 0; i < 8; ++i) {
            int m = mt * 16 + ((lane & 16) ? 8 : 0) + i;
            float v0 = vp[m * 4 + 0], v1 = vp[m * 4 + 1], v2 = vp[m * 4 + 2];
            const float* w0 = &sb[256 + jr * 3];
            const float* w1 = &sb[256 + (32 + jr) * 3];
            const float* w2 = &sb[256 + (64 + jr) * 3];
            ar[i]  += v0 * w0[0] + v1 * w0[1] + v2 * w0[2];
            az[i]  += v0 * w1[0] + v1 * w1[1] + v2 * w1[2];
            ain[i] += v0 * w2[0] + v1 * w2[1] + v2 * w2[2];
        }
        // x part: K = 512, B fragments stream from L2-resident packed w_ih0_x
        const u16* wx = wxp + (size_t)wg * NTW * 16 * 512;
#pragma unroll 4
        for (int ks = 0; ks < 16; ++ks) {
            v16bf a  = loadA(xt, 520, mt * 16, ks * 32);
            v16bf brf = loadB(wx + ((ct    ) * 16 + ks) * 512);
            v16bf bzf = loadB(wx + ((2 + ct) * 16 + ks) * 512);
            v16bf bnf = loadB(wx + ((4 + ct) * 16 + ks) * 512);
            ar  = wmma_bf16(a, brf, ar);
            az  = wmma_bf16(a, bzf, az);
            ain = wmma_bf16(a, bnf, ain);
        }
        // h part: K = 256, B fragments from LDS (w_hh0)
#pragma unroll 4
        for (int ks = 0; ks < 8; ++ks) {
            v16bf a  = loadA(h0f, 264, mt * 16, ks * 32);
            v16bf brf = loadB(wlds + ((ct    ) * 8 + ks) * 512);
            v16bf bzf = loadB(wlds + ((2 + ct) * 8 + ks) * 512);
            v16bf bnf = loadB(wlds + ((4 + ct) * 8 + ks) * 512);
            ar  = wmma_bf16(a, brf, ar);
            az  = wmma_bf16(a, bzf, az);
            ahn = wmma_bf16(a, bnf, ahn);
        }
        // GRU update, write own slice to global exchange buffer
#pragma unroll
        for (int i = 0; i < 8; ++i) {
            int m = mt * 16 + ((lane & 16) ? 8 : 0) + i;
            float r = sigm(ar[i]), z = sigm(az[i]);
            float n = fast_tanh(ain[i] + r * ahn[i]);
            float hp = bf2f(h0f[m * 264 + cg]);
            h0buf[m * NCH + cg] = f2bf((1.f - z) * n + z * hp);
        }
        gbar(bar, NWG * (2 * t + 1));
        // refresh full h0 in LDS: async DMA, no VGPR round trip
        for (int c = tid; c < BATCH * 32; c += 256) {    // 2048 x 16B
            int m = c >> 5, iw = c & 31;
            async_g2l_b128(l_h0 + m * 528 + iw * 16,
                           (const char*)h0buf + m * 512 + iw * 16);
        }
        wait_async0();
        __syncthreads();

        // ============================ layer 1 ==============================
        ar  = splat8(sb[128 + jr]);
        az  = splat8(sb[160 + jr]);
        ain = splat8(sb[192 + jr]);
        ahn = splat8(sb[224 + jr]);
#pragma unroll 4
        for (int ks = 0; ks < 8; ++ks) {           // gi1 = h0_new @ w_ih1^T
            v16bf a  = loadA(h0f, 264, mt * 16, ks * 32);
            v16bf brf = loadB(wlds + 24576 + ((ct    ) * 8 + ks) * 512);
            v16bf bzf = loadB(wlds + 24576 + ((2 + ct) * 8 + ks) * 512);
            v16bf bnf = loadB(wlds + 24576 + ((4 + ct) * 8 + ks) * 512);
            ar  = wmma_bf16(a, brf, ar);
            az  = wmma_bf16(a, bzf, az);
            ain = wmma_bf16(a, bnf, ain);
        }
#pragma unroll 4
        for (int ks = 0; ks < 8; ++ks) {           // gh1 = h1_prev @ w_hh1^T
            v16bf a  = loadA(h1f, 264, mt * 16, ks * 32);
            v16bf brf = loadB(wlds + 49152 + ((ct    ) * 8 + ks) * 512);
            v16bf bzf = loadB(wlds + 49152 + ((2 + ct) * 8 + ks) * 512);
            v16bf bnf = loadB(wlds + 49152 + ((4 + ct) * 8 + ks) * 512);
            ar  = wmma_bf16(a, brf, ar);
            az  = wmma_bf16(a, bzf, az);
            ahn = wmma_bf16(a, bnf, ahn);
        }
#pragma unroll
        for (int i = 0; i < 8; ++i) {
            int m = mt * 16 + ((lane & 16) ? 8 : 0) + i;
            float r = sigm(ar[i]), z = sigm(az[i]);
            float n = fast_tanh(ain[i] + r * ahn[i]);
            float hp = bf2f(h1f[m * 264 + cg]);
            h1buf[m * NCH + cg] = f2bf((1.f - z) * n + z * hp);
        }
        gbar(bar, NWG * (2 * t + 2));
        for (int c = tid; c < BATCH * 32; c += 256) {    // 2048 x 16B
            int m = c >> 5, iw = c & 31;
            async_g2l_b128(l_h1 + m * 528 + iw * 16,
                           (const char*)h1buf + m * 512 + iw * 16);
        }
        wait_async0();
        __syncthreads();

        // ---- projection + sigmoid; every WG redundantly (avoids a barrier)
        if (tid < BATCH * ODIM) {
            int m = tid & 63, o = tid >> 6;
            float acc = spw[768 + o];
            const float* prow = &spw[o * NCH];
            for (int c = 0; c < NCH; ++c) acc += bf2f(h1f[m * 264 + c]) * prow[c];
            float v = sigm(acc);
            vp[m * 4 + o] = v;
            if (wg == 0) out[((size_t)m * SEQ + t) * ODIM + o] = v;
        }
        __syncthreads();
    }
}

// ---------------------------------------------------------------------------
extern "C" void kernel_launch(void* const* d_in, const int* in_sizes, int n_in,
                              void* d_out, int out_size, void* d_ws, size_t ws_size,
                              hipStream_t stream) {
    (void)in_sizes; (void)n_in; (void)out_size; (void)ws_size;
    const float* x       = (const float*)d_in[0];
    const float* embed_w = (const float*)d_in[1];
    const float* embed_b = (const float*)d_in[2];
    const float* w_ih0   = (const float*)d_in[3];
    const float* w_hh0   = (const float*)d_in[4];
    const float* b_ih0   = (const float*)d_in[5];
    const float* b_hh0   = (const float*)d_in[6];
    const float* w_ih1   = (const float*)d_in[7];
    const float* w_hh1   = (const float*)d_in[8];
    const float* b_ih1   = (const float*)d_in[9];
    const float* b_hh1   = (const float*)d_in[10];
    const float* proj_w  = (const float*)d_in[11];
    const float* proj_b  = (const float*)d_in[12];

    char* ws = (char*)d_ws;
    u16*   wxp = (u16*)(ws + WS_W0X);
    u16*   whp = (u16*)(ws + WS_WH);
    float* ba0 = (float*)(ws + WS_BA0);
    float* bh0 = (float*)(ws + WS_BH0);
    float* ba1 = (float*)(ws + WS_BA1);
    float* bh1 = (float*)(ws + WS_BH1);
    float* wc  = (float*)(ws + WS_WC);
    float* pw  = (float*)(ws + WS_PW);
    float* pb  = (float*)(ws + WS_PB);
    u16*   h0b = (u16*)(ws + WS_H0);
    u16*   h1b = (u16*)(ws + WS_H1);
    int*   bar = (int*)(ws + WS_BAR);

    vad_init<<<1, 64, 0, stream>>>(bar);
    vad_pack_small<<<3, 256, 0, stream>>>(w_ih0, embed_w, embed_b, b_ih0, b_hh0,
                                          b_ih1, b_hh1, proj_w, proj_b,
                                          ba0, bh0, ba1, bh1, wc, pw, pb);
    vad_pack_wx<<<3072, 256, 0, stream>>>(w_ih0, wxp);
    vad_pack_wh<<<2304, 256, 0, stream>>>(w_hh0, w_ih1, w_hh1, whp);
    vad_rnn<<<NWG, 256, SMEM_BYTES, stream>>>(x, wxp, whp, ba0, bh0, ba1, bh1,
                                              wc, pw, pb, h0b, h1b, bar,
                                              (float*)d_out);
}
